// TransformerBlock_22445499089701
// MI455X (gfx1250) — matrix operand order repaired and checked by reading, not compiled, not run on hardware
//
#include <hip/hip_runtime.h>
#include <hip/hip_bf16.h>
#include <math.h>

// ---------------- Shapes (fixed by the reference) ----------------
#define EMBED   1024
#define HEADS   16
#define HEAD_DIM 64
#define FF_DIM  4096
#define BATCH   2
#define SEQ     2048
#define E3      (3 * EMBED)       // 3072
#define M_ROWS  (BATCH * SEQ)     // 4096

// ---------------- WMMA types / helpers ----------------
typedef float  v8f   __attribute__((ext_vector_type(8)));
typedef __bf16 v16bf __attribute__((ext_vector_type(16)));

union AFrag {
    unsigned u[8];
    v16bf    v;
};

static __device__ inline v8f wmma_bf16(v16bf a, v16bf b, v8f c) {
    // (neg_a, A, neg_b, B, c_mod, C, reuse_a, reuse_b)
    return __builtin_amdgcn_wmma_f32_16x16x32_bf16(
        false, a, false, b, (short)0, c, false, false);
}

// f32 -> bf16 bits, round-to-nearest-even
static __device__ inline unsigned short f2bf(float f) {
    unsigned u = __float_as_uint(f);
    u += 0x7fffu + ((u >> 16) & 1u);
    return (unsigned short)(u >> 16);
}

// xor-reduce max/sum across the 16 lanes that share a C-matrix row.
// ds_swizzle group-of-32 encoding: offset = and_mask | (or<<5) | (xor<<10)
static __device__ inline float xmax16(float v) {
    v = fmaxf(v, __int_as_float(__builtin_amdgcn_ds_swizzle(__float_as_int(v), 0x041f))); // xor 1
    v = fmaxf(v, __int_as_float(__builtin_amdgcn_ds_swizzle(__float_as_int(v), 0x081f))); // xor 2
    v = fmaxf(v, __int_as_float(__builtin_amdgcn_ds_swizzle(__float_as_int(v), 0x101f))); // xor 4
    v = fmaxf(v, __int_as_float(__builtin_amdgcn_ds_swizzle(__float_as_int(v), 0x201f))); // xor 8
    return v;
}
static __device__ inline float xsum16(float v) {
    v += __int_as_float(__builtin_amdgcn_ds_swizzle(__float_as_int(v), 0x041f));
    v += __int_as_float(__builtin_amdgcn_ds_swizzle(__float_as_int(v), 0x081f));
    v += __int_as_float(__builtin_amdgcn_ds_swizzle(__float_as_int(v), 0x101f));
    v += __int_as_float(__builtin_amdgcn_ds_swizzle(__float_as_int(v), 0x201f));
    return v;
}

// ---------------- Kernel 1: weight transpose + f32->bf16 ----------------
// W is [K][N] row-major f32; Wt is [N][K] bf16 so B-fragments load contiguous
// K-pairs as dwords.
__global__ __launch_bounds__(256) void transpose_bf16_kernel(
    const float* __restrict__ W, unsigned short* __restrict__ Wt, int K, int N) {
    long idx = (long)blockIdx.x * 256 + threadIdx.x;
    long total = (long)K * N;
    if (idx < total) {
        int k = (int)(idx / N);
        int n = (int)(idx % N);
        Wt[(long)n * K + k] = f2bf(W[idx]);
    }
}

// ---------------- Kernel 1b: V head-transpose ----------------
// qkv bf16 [B*S][3E]  ->  vT bf16 [B*H][HEAD_DIM][S]  so attention V-fragments
// (inner dim = key) load contiguous key-pairs as dwords.
__global__ __launch_bounds__(256) void vtrans_kernel(
    const unsigned short* __restrict__ qkv, unsigned short* __restrict__ vT) {
    long idx = (long)blockIdx.x * 256 + threadIdx.x;   // over B*H*HEAD_DIM*S
    int  s   = (int)(idx & (SEQ - 1));
    long r   = idx >> 11;                              // / SEQ
    int  d   = (int)(r & (HEAD_DIM - 1));
    long bh  = r >> 6;                                 // / HEAD_DIM
    int  b   = (int)(bh >> 4);                         // HEADS == 16
    int  h   = (int)(bh & 15);
    vT[idx] = qkv[(long)(b * SEQ + s) * E3 + 2 * EMBED + h * HEAD_DIM + d];
}

// ---------------- Kernel 2: LayerNorm (f32 in, bf16 out) ----------------
__global__ __launch_bounds__(256) void layernorm_bf16_kernel(
    const float* __restrict__ X, const float* __restrict__ gamma,
    const float* __restrict__ beta, unsigned short* __restrict__ H) {
    __shared__ float s_sum[256];
    __shared__ float s_sq[256];
    int row = blockIdx.x;
    int tid = threadIdx.x;
    const float* xr = X + (long)row * EMBED;

    float x0 = xr[tid * 4 + 0], x1 = xr[tid * 4 + 1];
    float x2 = xr[tid * 4 + 2], x3 = xr[tid * 4 + 3];
    s_sum[tid] = x0 + x1 + x2 + x3;
    s_sq[tid]  = x0 * x0 + x1 * x1 + x2 * x2 + x3 * x3;
    __syncthreads();
    for (int s = 128; s > 0; s >>= 1) {
        if (tid < s) { s_sum[tid] += s_sum[tid + s]; s_sq[tid] += s_sq[tid + s]; }
        __syncthreads();
    }
    float mean = s_sum[0] * (1.0f / EMBED);
    float var  = s_sq[0] * (1.0f / EMBED) - mean * mean;
    float rstd = rsqrtf(var + 1e-5f);

    unsigned short* hr = H + (long)row * EMBED;
    float xs[4] = {x0, x1, x2, x3};
#pragma unroll
    for (int c = 0; c < 4; ++c) {
        int col = tid * 4 + c;
        float hv = gamma[col] * (xs[c] - mean) * rstd + beta[col];
        hr[col] = f2bf(hv);
    }
}

// ---------------- Kernel 3: bf16 WMMA GEMM with fused epilogue ----------------
// C[M][N] = A[M][K] @ Bt[N][K]^T  (+bias, optional GELU, optional residual)
// Block: 256 threads = 8 waves in a 2(M) x 4(N) arrangement.
// Wave tile: 64x32 (4x2 WMMA tiles -> 8 WMMAs per 6 fragment loads per K-step).
// Block tile: 128x128. K step: 32.
__global__ __launch_bounds__(256) void gemm_bf16_wmma_kernel(
    const unsigned short* __restrict__ A, const unsigned short* __restrict__ Bt,
    const float* __restrict__ bias, const float* __restrict__ resid,
    float* __restrict__ outF, unsigned short* __restrict__ outBF,
    int M, int N, int K, int actGelu) {
    int tid  = threadIdx.x;
    int w    = tid >> 5;
    int lane = tid & 31;
    int wm   = w & 1;        // 0..1 along M
    int wn   = w >> 1;       // 0..3 along N
    int ml   = lane & 15;
    int half = lane >> 4;

    int rowBase = blockIdx.y * 128 + wm * 64;
    int colBase = blockIdx.x * 128 + wn * 32;

    v8f acc[4][2] = {};

    for (int k0 = 0; k0 < K; k0 += 32) {
        AFrag af[4], bf[2];
#pragma unroll
        for (int i = 0; i < 4; ++i) {
            const unsigned short* ap = A + (long)(rowBase + i * 16 + ml) * K + k0;
#pragma unroll
            for (int v = 0; v < 8; ++v) {
                int kk = ((v >> 2) << 4) + (half << 3) + ((v & 3) << 1);
                af[i].u[v] = *(const unsigned*)(ap + kk);
            }
        }
#pragma unroll
        for (int j = 0; j < 2; ++j) {
            const unsigned short* bp = Bt + (long)(colBase + j * 16 + ml) * K + k0;
#pragma unroll
            for (int v = 0; v < 8; ++v) {
                int kk = ((v >> 2) << 4) + (half << 3) + ((v & 3) << 1);
                bf[j].u[v] = *(const unsigned*)(bp + kk);
            }
        }
#pragma unroll
        for (int i = 0; i < 4; ++i)
#pragma unroll
            for (int j = 0; j < 2; ++j)
                acc[i][j] = wmma_bf16(af[i].v, bf[j].v, acc[i][j]);
    }

#pragma unroll
    for (int i = 0; i < 4; ++i) {
#pragma unroll
        for (int j = 0; j < 2; ++j) {
            int gc = colBase + j * 16 + ml;
            float bval = bias ? bias[gc] : 0.0f;
#pragma unroll
            for (int r = 0; r < 8; ++r) {
                int gr = rowBase + i * 16 + r + half * 8;
                float v = acc[i][j][r] + bval;
                if (actGelu) {
                    float x = v;
                    v = 0.5f * x * (1.0f + tanhf(0.7978845608f * (x + 0.044715f * x * x * x)));
                }
                long idx = (long)gr * N + gc;
                if (resid) v += resid[idx];
                if (outF)  outF[idx] = v;
                if (outBF) outBF[idx] = f2bf(v);
            }
        }
    }
}

// ---------------- Kernel 4: causal flash attention ----------------
// qkv: [B*S][3E] bf16 (bias fused by GEMM). vT: [B*H][64][S] bf16.
// out: [B*S][E] bf16 (heads merged).
// Block: 128 threads (4 waves); each wave owns a 16-row query tile.
// grid = (SEQ/64, BATCH*HEADS).
__global__ __launch_bounds__(128) void attn_flash_kernel(
    const unsigned short* __restrict__ qkv, const unsigned short* __restrict__ vT,
    unsigned short* __restrict__ out) {
    __shared__ unsigned short pbuf[4][16 * 32];   // per-wave P staging (C->A relayout)

    int w    = threadIdx.x >> 5;
    int lane = threadIdx.x & 31;
    int ml   = lane & 15;
    int half = lane >> 4;

    int bh = blockIdx.y;
    int b  = bh >> 4;            // HEADS == 16
    int h  = bh & 15;
    int q0 = blockIdx.x * 64 + w * 16;

    const unsigned short* qbase = qkv + (long)(b * SEQ) * E3 + h * HEAD_DIM;
    const unsigned short* kbase = qbase + EMBED;
    const unsigned short* vbase = vT + (long)bh * HEAD_DIM * SEQ;
    unsigned short* pb = pbuf[w];

    // Q fragments (row = q0+ml, d in [t*32, t*32+32))
    AFrag aq[2];
#pragma unroll
    for (int t = 0; t < 2; ++t) {
        const unsigned short* qp = qbase + (long)(q0 + ml) * E3 + t * 32;
#pragma unroll
        for (int v = 0; v < 8; ++v) {
            int kk = ((v >> 2) << 4) + (half << 3) + ((v & 3) << 1);
            aq[t].u[v] = *(const unsigned*)(qp + kk);
        }
    }

    v8f o[4] = {};
    float mrow[8], lrow[8];
#pragma unroll
    for (int r = 0; r < 8; ++r) { mrow[r] = -1e30f; lrow[r] = 0.0f; }

    for (int kb = 0; kb < q0 + 16; kb += 32) {
        // ---- scores S = Q @ K^T for 32 keys (two 16-col subtiles) ----
        v8f s[2] = {};
#pragma unroll
        for (int nb = 0; nb < 2; ++nb) {
            const unsigned short* kp = kbase + (long)(kb + nb * 16 + ml) * E3;
            AFrag bk0, bk1;
#pragma unroll
            for (int v = 0; v < 8; ++v) {
                int kk = ((v >> 2) << 4) + (half << 3) + ((v & 3) << 1);
                bk0.u[v] = *(const unsigned*)(kp + kk);
                bk1.u[v] = *(const unsigned*)(kp + 32 + kk);
            }
            s[nb] = wmma_bf16(aq[0].v, bk0.v, s[nb]);
            s[nb] = wmma_bf16(aq[1].v, bk1.v, s[nb]);
        }

        // ---- scale, causal mask, streaming softmax update ----
#pragma unroll
        for (int r = 0; r < 8; ++r) {
            int qrow = q0 + r + half * 8;
            float a0 = s[0][r] * 0.125f;                 // 1/sqrt(64)
            float a1 = s[1][r] * 0.125f;
            if (kb + ml      > qrow) a0 = -1e30f;
            if (kb + 16 + ml > qrow) a1 = -1e30f;
            float rmax = xmax16(fmaxf(a0, a1));
            float mnew = fmaxf(mrow[r], rmax);
            float scale = expf(mrow[r] - mnew);
            float p0 = expf(a0 - mnew);
            float p1 = expf(a1 - mnew);
            lrow[r] = lrow[r] * scale + xsum16(p0 + p1);
            mrow[r] = mnew;
#pragma unroll
            for (int j = 0; j < 4; ++j) o[j][r] *= scale;
            int rl = r + half * 8;
            pb[rl * 32 + ml]      = f2bf(p0);
            pb[rl * 32 + 16 + ml] = f2bf(p1);
        }
        asm volatile("s_wait_dscnt 0x0" ::: "memory");

        // ---- P (16x32) as A fragment from LDS ----
        AFrag ap;
#pragma unroll
        for (int v = 0; v < 8; ++v) {
            int kk = ((v >> 2) << 4) + (half << 3) + ((v & 3) << 1);
            ap.u[v] = *(const unsigned*)(pb + ml * 32 + kk);
        }
        asm volatile("s_wait_dscnt 0x0" ::: "memory");

        // ---- O += P @ V  (V block: 32 keys x 64 d, 4 column tiles) ----
        // vT layout [d][S]: lane ml -> column d = j*16+ml, key pairs contiguous.
#pragma unroll
        for (int j = 0; j < 4; ++j) {
            AFrag bv;
            const unsigned short* vp = vbase + (long)(j * 16 + ml) * SEQ + kb;
#pragma unroll
            for (int v = 0; v < 8; ++v) {
                int kk = ((v >> 2) << 4) + (half << 3) + ((v & 3) << 1);
                bv.u[v] = *(const unsigned*)(vp + kk);
            }
            o[j] = wmma_bf16(ap.v, bv.v, o[j]);
        }
    }

    // ---- finalize: divide by l, write bf16 merged-head output ----
#pragma unroll
    for (int j = 0; j < 4; ++j) {
#pragma unroll
        for (int r = 0; r < 8; ++r) {
            int gr = q0 + r + half * 8;
            float v = o[j][r] / lrow[r];
            out[(long)(b * SEQ + gr) * EMBED + h * HEAD_DIM + j * 16 + ml] = f2bf(v);
        }
    }
}

// ---------------- Host-side orchestration ----------------
extern "C" void kernel_launch(void* const* d_in, const int* in_sizes, int n_in,
                              void* d_out, int out_size, void* d_ws, size_t ws_size,
                              hipStream_t stream) {
    const float* x      = (const float*)d_in[0];
    const float* ln1_g  = (const float*)d_in[1];
    const float* ln1_b  = (const float*)d_in[2];
    const float* w_qkv  = (const float*)d_in[3];
    const float* b_qkv  = (const float*)d_in[4];
    const float* w_out  = (const float*)d_in[5];
    const float* b_out  = (const float*)d_in[6];
    const float* ln2_g  = (const float*)d_in[7];
    const float* ln2_b  = (const float*)d_in[8];
    const float* w_fc1  = (const float*)d_in[9];
    const float* b_fc1  = (const float*)d_in[10];
    const float* w_fc2  = (const float*)d_in[11];
    const float* b_fc2  = (const float*)d_in[12];
    float* out = (float*)d_out;

    // Workspace carve-up (256B aligned)
    char* ws = (char*)d_ws;
    size_t off = 0;
    auto carve = [&](size_t bytes) {
        char* p = ws + off;
        off = (off + bytes + 255) & ~(size_t)255;
        return p;
    };
    unsigned short* wqkvT  = (unsigned short*)carve((size_t)E3 * EMBED * 2);
    unsigned short* woutT  = (unsigned short*)carve((size_t)EMBED * EMBED * 2);
    unsigned short* wfc1T  = (unsigned short*)carve((size_t)FF_DIM * EMBED * 2);
    unsigned short* wfc2T  = (unsigned short*)carve((size_t)EMBED * FF_DIM * 2);
    unsigned short* h_bf   = (unsigned short*)carve((size_t)M_ROWS * EMBED * 2);
    unsigned short* qkv_bf = (unsigned short*)carve((size_t)M_ROWS * E3 * 2);
    unsigned short* vT     = (unsigned short*)carve((size_t)BATCH * HEADS * HEAD_DIM * SEQ * 2);
    unsigned short* attn_bf= (unsigned short*)carve((size_t)M_ROWS * EMBED * 2);
    float*          x1     = (float*)carve((size_t)M_ROWS * EMBED * 4);
    unsigned short* ff1_bf = (unsigned short*)carve((size_t)M_ROWS * FF_DIM * 2);
    (void)ws_size; (void)in_sizes; (void)n_in; (void)out_size;

    // 1) Weight transpose+convert
    {
        long n;
        n = (long)EMBED * E3;
        transpose_bf16_kernel<<<dim3((n + 255) / 256), 256, 0, stream>>>(w_qkv, wqkvT, EMBED, E3);
        n = (long)EMBED * EMBED;
        transpose_bf16_kernel<<<dim3((n + 255) / 256), 256, 0, stream>>>(w_out, woutT, EMBED, EMBED);
        n = (long)EMBED * FF_DIM;
        transpose_bf16_kernel<<<dim3((n + 255) / 256), 256, 0, stream>>>(w_fc1, wfc1T, EMBED, FF_DIM);
        n = (long)FF_DIM * EMBED;
        transpose_bf16_kernel<<<dim3((n + 255) / 256), 256, 0, stream>>>(w_fc2, wfc2T, FF_DIM, EMBED);
    }

    // 2) LN1: x -> h_bf
    layernorm_bf16_kernel<<<dim3(M_ROWS), 256, 0, stream>>>(x, ln1_g, ln1_b, h_bf);

    // 3) QKV GEMM: qkv_bf = h_bf @ w_qkv + b_qkv  [4096 x 3072]
    gemm_bf16_wmma_kernel<<<dim3(E3 / 128, M_ROWS / 128), 256, 0, stream>>>(
        h_bf, wqkvT, b_qkv, nullptr, nullptr, qkv_bf, M_ROWS, E3, EMBED, 0);

    // 3b) V head-transpose for coalesced P@V fragments
    {
        long n = (long)BATCH * HEADS * HEAD_DIM * SEQ;
        vtrans_kernel<<<dim3((n + 255) / 256), 256, 0, stream>>>(qkv_bf, vT);
    }

    // 4) Causal flash attention: qkv_bf (+vT) -> attn_bf [4096 x 1024]
    attn_flash_kernel<<<dim3(SEQ / 64, BATCH * HEADS), 128, 0, stream>>>(qkv_bf, vT, attn_bf);

    // 5) Out projection + residual: x1 = x + attn_bf @ w_out + b_out
    gemm_bf16_wmma_kernel<<<dim3(EMBED / 128, M_ROWS / 128), 256, 0, stream>>>(
        attn_bf, woutT, b_out, x, x1, nullptr, M_ROWS, EMBED, EMBED, 0);

    // 6) LN2: x1 -> h_bf (reuse)
    layernorm_bf16_kernel<<<dim3(M_ROWS), 256, 0, stream>>>(x1, ln2_g, ln2_b, h_bf);

    // 7) FC1 + GELU: ff1_bf = gelu(h_bf @ w_fc1 + b_fc1) [4096 x 4096]
    gemm_bf16_wmma_kernel<<<dim3(FF_DIM / 128, M_ROWS / 128), 256, 0, stream>>>(
        h_bf, wfc1T, b_fc1, nullptr, nullptr, ff1_bf, M_ROWS, FF_DIM, EMBED, 1);

    // 8) FC2 + residual: out = x1 + ff1_bf @ w_fc2 + b_fc2 [4096 x 1024]
    gemm_bf16_wmma_kernel<<<dim3(EMBED / 128, M_ROWS / 128), 256, 0, stream>>>(
        ff1_bf, wfc2T, b_fc2, x1, out, nullptr, M_ROWS, EMBED, FF_DIM, 0);
}
